// TransformerModel_69123203661902
// MI455X (gfx1250) — matrix operand order repaired and checked by reading, not compiled, not run on hardware
//
#include <hip/hip_runtime.h>
#include <cmath>

typedef __attribute__((ext_vector_type(16))) _Float16     v16h;
typedef __attribute__((ext_vector_type(8)))  float        v8f;
typedef __attribute__((ext_vector_type(4)))  unsigned int u32x4;
typedef __attribute__((ext_vector_type(2)))  unsigned int u32x2;
typedef __attribute__((ext_vector_type(4)))  float        f32x4;

#define LL 6
#define DD 512
#define DE 2048
#define HH 8
#define HD 64
#define NB 8
#define SS 1024
#define MM (NB * SS)
#define PB 40  // padded row stride (halves) for transposed LDS tiles: 80 B
               // keeps ds_load_b128 16B-aligned, spreads banks (gcd(20,64)=4)

union AF { v16h v; u32x4 q[2]; };
union H4 { u32x2 u; _Float16 h[4]; };
union H8 { u32x4 q; _Float16 h[8]; };

static __device__ __forceinline__ v8f wmma_f16(v16h a, v16h b, v8f c) {
  // D(f32 16x16) = A(f16 16x32) * B(f16 32x16) + C
  return __builtin_amdgcn_wmma_f32_16x16x32_f16(false, a, false, b, (short)0, c,
                                                false, false);
}

// butterfly reductions across the 16 lanes of each half-wave (bits 0..3 only,
// so lanes 0-15 and 16-31 reduce independently -> matches WMMA C-layout rows)
static __device__ __forceinline__ float redmax16(float x) {
  x = fmaxf(x, __shfl_xor(x, 1, 32));
  x = fmaxf(x, __shfl_xor(x, 2, 32));
  x = fmaxf(x, __shfl_xor(x, 4, 32));
  x = fmaxf(x, __shfl_xor(x, 8, 32));
  return x;
}
static __device__ __forceinline__ float redsum16(float x) {
  x += __shfl_xor(x, 1, 32);
  x += __shfl_xor(x, 2, 32);
  x += __shfl_xor(x, 4, 32);
  x += __shfl_xor(x, 8, 32);
  return x;
}

// ---------------------------------------------------------------- convert
__global__ void __launch_bounds__(128)
cvt_kernel(const float* __restrict__ src, _Float16* __restrict__ dst, int n4) {
  int i = blockIdx.x * blockDim.x + threadIdx.x;
  if (i >= n4) return;
  f32x4 v = ((const f32x4*)src)[i];
  H4 r;
#pragma unroll
  for (int j = 0; j < 4; ++j) r.h[j] = (_Float16)v[j];
  ((u32x2*)dst)[i] = r.u;
}

// ---------------------------------------------------------------- layernorm
__global__ void __launch_bounds__(128)
ln_kernel(const float* __restrict__ X, const float* __restrict__ gam,
          const float* __restrict__ bet, _Float16* __restrict__ out) {
  const int row = blockIdx.x, tid = threadIdx.x;
  const int lane = tid & 31, w = tid >> 5;
  f32x4 x = ((const f32x4*)(X + (size_t)row * DD))[tid];
  float s = x[0] + x[1] + x[2] + x[3];
  float q = x[0] * x[0] + x[1] * x[1] + x[2] * x[2] + x[3] * x[3];
#pragma unroll
  for (int m = 1; m < 32; m <<= 1) {
    s += __shfl_xor(s, m, 32);
    q += __shfl_xor(q, m, 32);
  }
  __shared__ float ss[4], sq[4];
  if (lane == 0) { ss[w] = s; sq[w] = q; }
  __syncthreads();
  s = ss[0] + ss[1] + ss[2] + ss[3];
  q = sq[0] + sq[1] + sq[2] + sq[3];
  float mu  = s * (1.0f / DD);
  float var = q * (1.0f / DD) - mu * mu;
  float rs  = rsqrtf(var + 1e-3f);
  f32x4 g = ((const f32x4*)gam)[tid];
  f32x4 b = ((const f32x4*)bet)[tid];
  H4 r;
#pragma unroll
  for (int j = 0; j < 4; ++j) r.h[j] = (_Float16)((x[j] - mu) * rs * g[j] + b[j]);
  ((u32x2*)(out + (size_t)row * DD))[tid] = r.u;
}

// ---------------------------------------------------------------- WMMA GEMM
// C[M,N] = A[M,K] @ B[K,N]  (A,B f16 row-major), epilogue per MODE:
//   0: f16 store    1: +bias, exact GELU, f16 store
//   2: +bias +res, f32 store   3: +res, f32 store   (res may alias outF)
// Block: 256 threads = 8 waves, 128x128 tile; wave tile 32x64 (2x4 WMMA C
// blocks, 8 v_wmma per K-step per wave). All tile copies are b128.
template <int MODE>
__global__ void __launch_bounds__(256)
gemm_kernel(const _Float16* __restrict__ A, const _Float16* __restrict__ Bm,
            const float* __restrict__ bias, const float* res, float* outF,
            _Float16* __restrict__ outH, int M, int N, int Kd) {
  __shared__ __align__(16) unsigned int As[128 * 16];  // 128 rows x 32 halves
  __shared__ __align__(16) _Float16 BsT[128 * PB];     // [ncol][k], padded
  const int tid = threadIdx.x;
  const int lane = tid & 31, w = tid >> 5;
  const int wr = w >> 1, wc = w & 1;                   // 4x2 wave grid
  const int n16 = lane & 15, half = lane >> 4;
  const int rowBlock = blockIdx.y * 128, colBlock = blockIdx.x * 128;

  v8f acc[2][4];
#pragma unroll
  for (int r = 0; r < 2; ++r)
#pragma unroll
    for (int t = 0; t < 4; ++t) acc[r][t] = 0;

  for (int k0 = 0; k0 < Kd; k0 += 32) {
#pragma unroll
    for (int j = 0; j < 2; ++j) {
      int i = tid + j * 256;               // 512 x 16B chunks each for A and B
      int row = i >> 2, c4 = i & 3;        // A: 128 rows x 4 x (8 halves)
      const _Float16* ag = A + (size_t)(rowBlock + row) * Kd + k0 + c4 * 8;
      ((u32x4*)As)[row * 4 + c4] = *(const u32x4*)ag;
      int kk = i >> 4, n8 = i & 15;        // B: 32 k-rows x 16 x (8 halves)
      const _Float16* bg = Bm + (size_t)(k0 + kk) * N + colBlock + n8 * 8;
      H8 hb;
      hb.q = *(const u32x4*)bg;
#pragma unroll
      for (int jj = 0; jj < 8; ++jj) BsT[(n8 * 8 + jj) * PB + kk] = hb.h[jj];
      if (k0 + 32 < Kd) {                  // speculative L2->WGP prefetch of
        __builtin_prefetch(ag + 32, 0, 0); // next K-step (global_prefetch_b8)
        __builtin_prefetch(bg + (size_t)32 * N, 0, 0);
      }
    }
    __syncthreads();
    // A fragments: rows m and m+16; halves k = half*16+e
    AF a0, a1;
    {
      const u32x4* ap = (const u32x4*)(As + (wr * 32 + n16) * 16);
      a0.q[0] = ap[2 * half];
      a0.q[1] = ap[2 * half + 1];
      const u32x4* ap2 = (const u32x4*)(As + (wr * 32 + 16 + n16) * 16);
      a1.q[0] = ap2[2 * half];
      a1.q[1] = ap2[2 * half + 1];
    }
#pragma unroll
    for (int t = 0; t < 4; ++t) {
      // B fragment: col n = lane%16; k = half*16 + e (contiguous in BsT)
      AF bf;
      const u32x4* bp =
          (const u32x4*)(BsT + (wc * 64 + t * 16 + n16) * PB + half * 16);
      bf.q[0] = bp[0];
      bf.q[1] = bp[1];
      acc[0][t] = wmma_f16(a0.v, bf.v, acc[0][t]);
      acc[1][t] = wmma_f16(a1.v, bf.v, acc[1][t]);
    }
    __syncthreads();
  }

#pragma unroll
  for (int r = 0; r < 2; ++r) {
    const int rbase = rowBlock + wr * 32 + r * 16 + half * 8;  // C row=v+8*half
#pragma unroll
    for (int t = 0; t < 4; ++t) {
      int cgl = colBlock + wc * 64 + t * 16 + n16;
#pragma unroll
      for (int v = 0; v < 8; ++v) {
        size_t idx = (size_t)(rbase + v) * N + cgl;
        float val = acc[r][t][v];
        if (MODE == 0) {
          outH[idx] = (_Float16)val;
        } else if (MODE == 1) {
          float z = val + bias[cgl];
          outH[idx] = (_Float16)(0.5f * z * (1.0f + erff(z * 0.70710678118f)));
        } else if (MODE == 2) {
          outF[idx] = res[idx] + val + bias[cgl];
        } else {
          outF[idx] = res[idx] + val;
        }
      }
    }
  }
}

// ---------------------------------------------------------------- flash attn
// one wave per (b, h, 16-query tile); online softmax over S=1024 keys
__global__ void __launch_bounds__(32)
attn_kernel(const _Float16* __restrict__ Q, const _Float16* __restrict__ Kp,
            const _Float16* __restrict__ V, _Float16* __restrict__ Z) {
  __shared__ __align__(16) _Float16 Plds[16 * PB];  // P block, padded rows
  __shared__ __align__(16) _Float16 VT[64 * PB];    // V^T: [d][key], padded
  const int lane = threadIdx.x;
  const int n = lane & 15, half = lane >> 4;
  const int qt = blockIdx.x, h = blockIdx.y, b = blockIdx.z;
  const int hd0 = h * HD;
  const int rq = b * SS + qt * 16;

  // Q fragments (A layout), d 0..31 and 32..63
  AF a0, a1;
  {
    const u32x4* qp = (const u32x4*)(Q + (size_t)(rq + n) * DD + hd0);
    a0.q[0] = qp[2 * half];
    a0.q[1] = qp[2 * half + 1];
    a1.q[0] = qp[4 + 2 * half];
    a1.q[1] = qp[5 + 2 * half];
  }

  float mrow[8], lrow[8];
  v8f o[4];
#pragma unroll
  for (int v = 0; v < 8; ++v) { mrow[v] = -3.0e38f; lrow[v] = 0.0f; }
#pragma unroll
  for (int t = 0; t < 4; ++t) o[t] = 0;

  for (int kb = 0; kb < SS; kb += 32) {
    // prefetch next key block's K/V rows (speculative, no arch effect)
    if (kb + 32 < SS) {
      __builtin_prefetch(Kp + (size_t)(b * SS + kb + 32 + lane) * DD + hd0, 0, 0);
      __builtin_prefetch(V + (size_t)(b * SS + kb + 32 + lane) * DD + hd0, 0, 0);
    }
    // stage V^T (32 keys x 64 d) : lane owns one key row
    {
      const _Float16* vp = V + (size_t)(b * SS + kb + lane) * DD + hd0;
#pragma unroll
      for (int c = 0; c < 8; ++c) {
        H8 u;
        u.q = ((const u32x4*)vp)[c];
#pragma unroll
        for (int j = 0; j < 8; ++j) VT[(c * 8 + j) * PB + lane] = u.h[j];
      }
    }
    // scores for 32 keys: two 16x16 C blocks
    v8f c0 = 0, c1 = 0;
    {
      const u32x4* kp = (const u32x4*)(Kp + (size_t)(b * SS + kb + n) * DD + hd0);
      AF f0, f1;
      f0.q[0] = kp[half * 2];
      f0.q[1] = kp[half * 2 + 1];
      f1.q[0] = kp[4 + half * 2];
      f1.q[1] = kp[5 + half * 2];
      c0 = wmma_f16(a0.v, f0.v, c0);
      c0 = wmma_f16(a1.v, f1.v, c0);
    }
    {
      const u32x4* kp =
          (const u32x4*)(Kp + (size_t)(b * SS + kb + 16 + n) * DD + hd0);
      AF f0, f1;
      f0.q[0] = kp[half * 2];
      f0.q[1] = kp[half * 2 + 1];
      f1.q[0] = kp[4 + half * 2];
      f1.q[1] = kp[5 + half * 2];
      c1 = wmma_f16(a0.v, f0.v, c1);
      c1 = wmma_f16(a1.v, f1.v, c1);
    }
    // online softmax (row = v + 8*half, col = n / n+16)
#pragma unroll
    for (int v = 0; v < 8; ++v) {
      float s0 = c0[v] * 0.125f, s1 = c1[v] * 0.125f;
      float mx = redmax16(fmaxf(s0, s1));
      float mn = fmaxf(mrow[v], mx);
      float al = __expf(mrow[v] - mn);
      float p0 = __expf(s0 - mn), p1 = __expf(s1 - mn);
      float ps = redsum16(p0 + p1);
      lrow[v] = lrow[v] * al + ps;
      mrow[v] = mn;
      o[0][v] *= al; o[1][v] *= al; o[2][v] *= al; o[3][v] *= al;
      Plds[(v + half * 8) * PB + n]      = (_Float16)p0;
      Plds[(v + half * 8) * PB + 16 + n] = (_Float16)p1;
    }
    __syncthreads();
    // P as A fragment (16x32), V^T rows as B fragments
    AF pa;
    const u32x4* pp = (const u32x4*)(Plds + n * PB);
    pa.q[0] = pp[2 * half];
    pa.q[1] = pp[2 * half + 1];
#pragma unroll
    for (int t = 0; t < 4; ++t) {
      AF vb;
      const u32x4* vt = (const u32x4*)(VT + (t * 16 + n) * PB + half * 16);
      vb.q[0] = vt[0];
      vb.q[1] = vt[1];
      o[t] = wmma_f16(pa.v, vb.v, o[t]);
    }
    __syncthreads();
  }
  // normalize and store Z (f16, heads concatenated)
#pragma unroll
  for (int t = 0; t < 4; ++t)
#pragma unroll
    for (int v = 0; v < 8; ++v)
      Z[(size_t)(rq + half * 8 + v) * DD + hd0 + t * 16 + n] =
          (_Float16)(o[t][v] / lrow[v]);
}

// ---------------------------------------------------------------- host
extern "C" void kernel_launch(void* const* d_in, const int* in_sizes, int n_in,
                              void* d_out, int out_size, void* d_ws,
                              size_t ws_size, hipStream_t stream) {
  (void)in_sizes; (void)n_in; (void)out_size; (void)ws_size;
  const float* x    = (const float*)d_in[0];
  const float* Wq   = (const float*)d_in[1];
  const float* Wk   = (const float*)d_in[2];
  const float* Wv   = (const float*)d_in[3];
  const float* Wo   = (const float*)d_in[4];
  const float* w1   = (const float*)d_in[5];
  const float* b1   = (const float*)d_in[6];
  const float* w2   = (const float*)d_in[7];
  const float* b2   = (const float*)d_in[8];
  const float* lnag = (const float*)d_in[9];
  const float* lnab = (const float*)d_in[10];
  const float* lnfg = (const float*)d_in[11];
  const float* lnfb = (const float*)d_in[12];

  float* enc = (float*)d_out;  // residual stream, f32, in-place across layers
  char* ws = (char*)d_ws;

  // workspace layout (bytes)
  _Float16* wq16 = (_Float16*)(ws + 0);
  _Float16* wk16 = (_Float16*)(ws + 524288);
  _Float16* wv16 = (_Float16*)(ws + 2 * 524288);
  _Float16* wo16 = (_Float16*)(ws + 3 * 524288);
  _Float16* w116 = (_Float16*)(ws + 4 * 524288);                 // 2 MiB
  _Float16* w216 = (_Float16*)(ws + 4 * 524288 + 2097152);
  char* act = ws + 4 * 524288 + 2 * 2097152;
  _Float16* xn16 = (_Float16*)(act);                 // LN out / Z (aliased)
  _Float16* q16  = (_Float16*)(act + 8388608);
  _Float16* k16  = (_Float16*)(act + 2 * 8388608);
  _Float16* v16  = (_Float16*)(act + 3 * 8388608);
  _Float16* h116 = (_Float16*)(act + 4 * 8388608);   // 32 MiB (8192x2048 f16)
  _Float16* z16  = xn16;

  hipMemcpyAsync(enc, x, (size_t)MM * DD * sizeof(float),
                 hipMemcpyDeviceToDevice, stream);

  const dim3 g512(DD / 128, MM / 128), g2048(DE / 128, MM / 128);
  const dim3 gattn(SS / 16, HH, NB);
  const int nW4 = DD * DD / 4, nF4 = DD * DE / 4;

  for (int i = 0; i < LL; ++i) {
    // per-layer weight conversion f32 -> f16
    cvt_kernel<<<nW4 / 128, 128, 0, stream>>>(Wq + (size_t)i * DD * DD, wq16, nW4);
    cvt_kernel<<<nW4 / 128, 128, 0, stream>>>(Wk + (size_t)i * DD * DD, wk16, nW4);
    cvt_kernel<<<nW4 / 128, 128, 0, stream>>>(Wv + (size_t)i * DD * DD, wv16, nW4);
    cvt_kernel<<<nW4 / 128, 128, 0, stream>>>(Wo + (size_t)i * DD * DD, wo16, nW4);
    cvt_kernel<<<nF4 / 128, 128, 0, stream>>>(w1 + (size_t)i * DD * DE, w116, nF4);
    cvt_kernel<<<nF4 / 128, 128, 0, stream>>>(w2 + (size_t)i * DE * DD, w216, nF4);

    // attention block
    ln_kernel<<<MM, 128, 0, stream>>>(enc, lnag + i * DD, lnab + i * DD, xn16);
    gemm_kernel<0><<<g512, 256, 0, stream>>>(xn16, wq16, nullptr, nullptr,
                                             nullptr, q16, MM, DD, DD);
    gemm_kernel<0><<<g512, 256, 0, stream>>>(xn16, wk16, nullptr, nullptr,
                                             nullptr, k16, MM, DD, DD);
    gemm_kernel<0><<<g512, 256, 0, stream>>>(xn16, wv16, nullptr, nullptr,
                                             nullptr, v16, MM, DD, DD);
    attn_kernel<<<gattn, 32, 0, stream>>>(q16, k16, v16, z16);
    gemm_kernel<3><<<g512, 256, 0, stream>>>(z16, wo16, nullptr, enc, enc,
                                             nullptr, MM, DD, DD);
    // FFN block
    ln_kernel<<<MM, 128, 0, stream>>>(enc, lnfg + i * DD, lnfb + i * DD, xn16);
    gemm_kernel<1><<<g2048, 256, 0, stream>>>(xn16, w116, b1 + (size_t)i * DE,
                                              nullptr, nullptr, h116, MM, DE, DD);
    gemm_kernel<2><<<g512, 256, 0, stream>>>(h116, w216, b2 + (size_t)i * DD,
                                             enc, enc, nullptr, MM, DD, DE);
  }
}
